// MultiHeadAttention_19198503813360
// MI455X (gfx1250) — compile-verified, matrix operand order checked by reading, not run on hardware
//
#include <hip/hip_runtime.h>

typedef __attribute__((ext_vector_type(16))) __bf16 v16bf;
typedef __attribute__((ext_vector_type(8)))  __bf16 v8bf;
typedef __attribute__((ext_vector_type(8)))  float  v8f;

#define B_   8
#define T_   16
#define D_   4096
#define M_   4096
#define HQ_  32
#define HK_  8
#define DK_  128
#define MULT_ 0.08838834764831845f
#define MAXA_ 30.0f

// 16-bit fragment K-index map (K=32): 4 chunks of 8;
// chunks 0/2 -> lanes 0-15, chunks 1/3 -> lanes 16-31 (low/high VGPR halves).
__device__ __forceinline__ int kmap(int j, int hi) {
  return ((j >> 3) << 4) + (hi << 3) + (j & 7);
}

// C[128,N] = A[128,K] @ (W.int8 * S.f32)[K,N].
// Block = 8 waves = 16x128 C tile; LDS-staged A (16x32) and dequant W (32x128,
// stored transposed so B-fragment reads are contiguous).
template<bool ABF16>
__global__ void __launch_bounds__(256) gemm_w8_kernel(
    const void* __restrict__ Av, const signed char* __restrict__ W,
    const float* __restrict__ S, float* __restrict__ C, int N, int K) {
  __shared__ __bf16 aT[16][32];     // [row][k]
  __shared__ __bf16 bTt[128][32];   // [n-local][k]  (transposed)
  const int tid  = threadIdx.x;
  const int lane = tid & 31, wv = tid >> 5;
  const int nblk = N >> 7;
  const int bm = blockIdx.x / nblk;
  const int bn = blockIdx.x % nblk;
  const int ln = lane & 15, hi = lane >> 4;
  const int n = bn * 128 + wv * 16 + ln;
  // staging assignments
  const int ar = tid >> 4, ak = (tid << 1) & 31;        // A: 2 elems/thread
  const int br = tid >> 3, bc = (tid & 7) * 16;         // B: 16 elems/thread
  v8f acc = {};
  for (int k0 = 0; k0 < K; k0 += 32) {
    // ---- stage A tile ----
    if constexpr (ABF16) {
      const __bf16* ap = (const __bf16*)Av + (size_t)(bm * 16 + ar) * K + k0 + ak;
      aT[ar][ak] = ap[0];
      aT[ar][ak + 1] = ap[1];
    } else {
      const float* ap = (const float*)Av + (size_t)(bm * 16 + ar) * K + k0 + ak;
      float2 f2 = *(const float2*)ap;
      aT[ar][ak] = (__bf16)f2.x;
      aT[ar][ak + 1] = (__bf16)f2.y;
    }
    // ---- stage dequantized W tile (coalesced int4 + 4x float4) ----
    {
      const signed char* wp = W + (size_t)(k0 + br) * N + bn * 128 + bc;
      const float*       sp = S + (size_t)(k0 + br) * N + bn * 128 + bc;
      int4 wraw = *(const int4*)wp;
      const signed char* wb = (const signed char*)&wraw;
      float4 s4[4];
#pragma unroll
      for (int i = 0; i < 4; ++i) s4[i] = *(const float4*)(sp + i * 4);
      const float* sfl = (const float*)s4;
#pragma unroll
      for (int i = 0; i < 16; ++i)
        bTt[bc + i][br] = (__bf16)((float)wb[i] * sfl[i]);
    }
    __syncthreads();
    // ---- fragments from LDS (contiguous 16B reads) ----
    v16bf a, b;
    {
      v8bf a0 = *(const v8bf*)&aT[ln][hi * 8];
      v8bf a1 = *(const v8bf*)&aT[ln][16 + hi * 8];
      v8bf b0 = *(const v8bf*)&bTt[wv * 16 + ln][hi * 8];
      v8bf b1 = *(const v8bf*)&bTt[wv * 16 + ln][16 + hi * 8];
#pragma unroll
      for (int j = 0; j < 8; ++j) {
        a[j] = a0[j]; a[8 + j] = a1[j];
        b[j] = b0[j]; b[8 + j] = b1[j];
      }
    }
    acc = __builtin_amdgcn_wmma_f32_16x16x32_bf16(false, a, false, b,
                                                  (short)0, acc, false, false);
    __syncthreads();
  }
#pragma unroll
  for (int r = 0; r < 8; ++r)
    C[(size_t)(bm * 16 + hi * 8 + r) * N + n] = acc[r];
}

// In-place RoPE on x:[B,T,H,DK], phase offset step[b].
__global__ void rope_kernel(float* __restrict__ x, const int* __restrict__ step,
                            int H) {
  int idx = blockIdx.x * blockDim.x + threadIdx.x;
  int j = idx & 63;
  int r = idx >> 6;
  int h = r % H; r /= H;
  int t = r & 15;
  int b = r >> 4;
  float invf = __powf(10000.f, -(float)(2 * j) * (1.0f / 128.f));
  float ph = (float)(t + step[b]) * invf;
  float s, c;
  __sincosf(ph, &s, &c);
  float* base = x + ((size_t)(b * 16 + t) * H + h) * DK_;
  float x1 = base[j], x2 = base[j + 64];
  base[j]      = x1 * c - x2 * s;
  base[j + 64] = x2 * c + x1 * s;
}

// Scatter the T new rows into the (already copied) caches; emit new_step.
__global__ void cache_update_kernel(const float* __restrict__ kb,
                                    const float* __restrict__ vb,
                                    const int* __restrict__ step,
                                    float* __restrict__ kc,
                                    float* __restrict__ vc,
                                    float* __restrict__ stepo) {
  int idx = blockIdx.x * blockDim.x + threadIdx.x;   // [B,T,HK,DK] flat
  int d = idx & 127;
  int h = (idx >> 7) & 7;
  int t = (idx >> 10) & 15;
  int b = idx >> 14;
  size_t dst = ((size_t)(b * M_ + step[b] + t) * HK_ + h) * DK_ + d;
  kc[dst] = kb[idx];
  vc[dst] = vb[idx];
  if (idx < B_) stepo[idx] = (float)(step[idx] + T_);
}

// Flash attention. Block = 4 waves, all sharing (b, kv-head h); each wave owns
// one q-head (16 t-rows x DK=128 accum). K/V tiles staged once per block.
__global__ void __launch_bounds__(128) attn_kernel(
    const float* __restrict__ qbuf, const float* __restrict__ kc,
    const float* __restrict__ vc, const unsigned char* __restrict__ mask,
    const int* __restrict__ step, __bf16* __restrict__ attnb) {
  __shared__ __bf16 kT[32][128];    // [key][d]
  __shared__ __bf16 vTt[128][32];   // [d][key]  (transposed)
  __shared__ __bf16 plds[4][16][32];
  const int tid  = threadIdx.x;
  const int lane = tid & 31;
  const int w    = tid >> 5;
  const int b    = blockIdx.x >> 3;        // 8 blocks per batch
  const int h    = blockIdx.x & 7;         // kv head, uniform per block
  const int qh   = h * 4 + w;              // q head for this wave
  const int ln = lane & 15, hi = lane >> 4;
  const int ns = step[b] + T_;
  // staging assignment: 4 threads per key row, 32 floats each
  const int sr = tid >> 2;
  const int sc = (tid & 3) * 32;

  v16bf qf[4];
#pragma unroll
  for (int c = 0; c < 4; ++c) {
    const float* qrow =
        &qbuf[((size_t)(b * 16 + ln) * HQ_ + qh) * DK_ + c * 32 + hi * 8];
    float4 q0 = *(const float4*)(qrow + 0);
    float4 q1 = *(const float4*)(qrow + 4);
    float4 q2 = *(const float4*)(qrow + 16);
    float4 q3 = *(const float4*)(qrow + 20);
    qf[c][0] = (__bf16)q0.x;  qf[c][1] = (__bf16)q0.y;
    qf[c][2] = (__bf16)q0.z;  qf[c][3] = (__bf16)q0.w;
    qf[c][4] = (__bf16)q1.x;  qf[c][5] = (__bf16)q1.y;
    qf[c][6] = (__bf16)q1.z;  qf[c][7] = (__bf16)q1.w;
    qf[c][8] = (__bf16)q2.x;  qf[c][9] = (__bf16)q2.y;
    qf[c][10] = (__bf16)q2.z; qf[c][11] = (__bf16)q2.w;
    qf[c][12] = (__bf16)q3.x; qf[c][13] = (__bf16)q3.y;
    qf[c][14] = (__bf16)q3.z; qf[c][15] = (__bf16)q3.w;
  }

  float mrun[8], lrun[8];
  v8f o[8];
#pragma unroll
  for (int r = 0; r < 8; ++r) { mrun[r] = -1e30f; lrun[r] = 0.f; }
#pragma unroll
  for (int dt = 0; dt < 8; ++dt)
#pragma unroll
    for (int r = 0; r < 8; ++r) o[dt][r] = 0.f;

  for (int jb = 0; jb < M_; jb += 32) {
    if (jb >= ns) break;              // uniform per block: later keys masked
    // ---- cooperative K/V tile staging (coalesced float4 rows) ----
    {
      const float* krow = &kc[((size_t)(b * M_ + jb + sr) * HK_ + h) * DK_ + sc];
      const float* vrow = &vc[((size_t)(b * M_ + jb + sr) * HK_ + h) * DK_ + sc];
#pragma unroll
      for (int i = 0; i < 8; ++i) {
        float4 kf = *(const float4*)(krow + i * 4);
        float4 vf4 = *(const float4*)(vrow + i * 4);
        kT[sr][sc + i * 4 + 0] = (__bf16)kf.x;
        kT[sr][sc + i * 4 + 1] = (__bf16)kf.y;
        kT[sr][sc + i * 4 + 2] = (__bf16)kf.z;
        kT[sr][sc + i * 4 + 3] = (__bf16)kf.w;
        vTt[sc + i * 4 + 0][sr] = (__bf16)vf4.x;
        vTt[sc + i * 4 + 1][sr] = (__bf16)vf4.y;
        vTt[sc + i * 4 + 2][sr] = (__bf16)vf4.z;
        vTt[sc + i * 4 + 3][sr] = (__bf16)vf4.w;
      }
    }
    __syncthreads();
    // ---- S = Q K^T over 32 keys ----
    v8f s0 = {}, s1 = {};
#pragma unroll
    for (int c = 0; c < 4; ++c) {
      v16bf k0f, k1f;
      v8bf t0 = *(const v8bf*)&kT[ln][c * 32 + hi * 8];
      v8bf t1 = *(const v8bf*)&kT[ln][c * 32 + 16 + hi * 8];
      v8bf u0 = *(const v8bf*)&kT[16 + ln][c * 32 + hi * 8];
      v8bf u1 = *(const v8bf*)&kT[16 + ln][c * 32 + 16 + hi * 8];
#pragma unroll
      for (int j = 0; j < 8; ++j) {
        k0f[j] = t0[j]; k0f[8 + j] = t1[j];
        k1f[j] = u0[j]; k1f[8 + j] = u1[j];
      }
      s0 = __builtin_amdgcn_wmma_f32_16x16x32_bf16(false, qf[c], false, k0f,
                                                   (short)0, s0, false, false);
      s1 = __builtin_amdgcn_wmma_f32_16x16x32_bf16(false, qf[c], false, k1f,
                                                   (short)0, s1, false, false);
    }
    // ---- tanh cap, mask, online softmax ----
    float al[8];
#pragma unroll
    for (int r = 0; r < 8; ++r) {
      int t = hi * 8 + r;
      int p0 = jb + ln, p1 = jb + 16 + ln;
      float x0 = MAXA_ * tanhf(s0[r] * (MULT_ / MAXA_));
      float x1 = MAXA_ * tanhf(s1[r] * (MULT_ / MAXA_));
      bool v0 = (p0 < ns) && mask[(size_t)(b * T_ + t) * M_ + p0];
      bool v1 = (p1 < ns) && mask[(size_t)(b * T_ + t) * M_ + p1];
      x0 = v0 ? x0 : -1e30f;
      x1 = v1 ? x1 : -1e30f;
      float rm = fmaxf(x0, x1);
      rm = fmaxf(rm, __shfl_xor(rm, 1, 32));
      rm = fmaxf(rm, __shfl_xor(rm, 2, 32));
      rm = fmaxf(rm, __shfl_xor(rm, 4, 32));
      rm = fmaxf(rm, __shfl_xor(rm, 8, 32));
      float mnew  = fmaxf(mrun[r], rm);
      float alpha = __expf(mrun[r] - mnew);
      float p0e = __expf(x0 - mnew);
      float p1e = __expf(x1 - mnew);
      float rs = p0e + p1e;
      rs += __shfl_xor(rs, 1, 32);
      rs += __shfl_xor(rs, 2, 32);
      rs += __shfl_xor(rs, 4, 32);
      rs += __shfl_xor(rs, 8, 32);
      lrun[r] = alpha * lrun[r] + rs;
      mrun[r] = mnew;
      al[r] = alpha;
      plds[w][t][ln]      = (__bf16)p0e;   // C-layout -> per-wave LDS tile
      plds[w][t][16 + ln] = (__bf16)p1e;
    }
#pragma unroll
    for (int dt = 0; dt < 8; ++dt)
#pragma unroll
      for (int r = 0; r < 8; ++r) o[dt][r] *= al[r];
    asm volatile("s_wait_dscnt 0x0" ::: "memory");   // P stores visible to wave
    v16bf pf;                                        // A-layout reload
    {
      v8bf p0v = *(const v8bf*)&plds[w][ln][hi * 8];
      v8bf p1v = *(const v8bf*)&plds[w][ln][16 + hi * 8];
#pragma unroll
      for (int j = 0; j < 8; ++j) { pf[j] = p0v[j]; pf[8 + j] = p1v[j]; }
    }
    asm volatile("s_wait_dscnt 0x0" ::: "memory");   // P loads before next store
    // ---- O += P V (V transposed in LDS -> contiguous fragment reads) ----
#pragma unroll
    for (int dt = 0; dt < 8; ++dt) {
      v16bf vfg;
      v8bf a0 = *(const v8bf*)&vTt[dt * 16 + ln][hi * 8];
      v8bf a1 = *(const v8bf*)&vTt[dt * 16 + ln][16 + hi * 8];
#pragma unroll
      for (int j = 0; j < 8; ++j) { vfg[j] = a0[j]; vfg[8 + j] = a1[j]; }
      o[dt] = __builtin_amdgcn_wmma_f32_16x16x32_bf16(false, pf, false, vfg,
                                                      (short)0, o[dt], false, false);
    }
    __syncthreads();   // all waves done with kT/vTt before next staging
  }
#pragma unroll
  for (int dt = 0; dt < 8; ++dt)
#pragma unroll
    for (int r = 0; r < 8; ++r) {
      float v = o[dt][r] / lrun[r];
      attnb[(size_t)(b * 16 + hi * 8 + r) * (HQ_ * DK_) + qh * DK_ + dt * 16 + ln] =
          (__bf16)v;
    }
}

extern "C" void kernel_launch(void* const* d_in, const int* in_sizes, int n_in,
                              void* d_out, int out_size, void* d_ws, size_t ws_size,
                              hipStream_t stream) {
  const float* query = (const float*)d_in[0];
  const float* key_  = (const float*)d_in[1];
  const float* value = (const float*)d_in[2];
  const unsigned char* mask = (const unsigned char*)d_in[3];
  const float* mem_k = (const float*)d_in[4];
  const float* mem_v = (const float*)d_in[5];
  const int*   step  = (const int*)d_in[6];
  const signed char* wq = (const signed char*)d_in[7];
  const float*       sq = (const float*)d_in[8];
  const signed char* wk = (const signed char*)d_in[9];
  const float*       sk = (const float*)d_in[10];
  const signed char* wv = (const signed char*)d_in[11];
  const float*       sv = (const float*)d_in[12];
  const signed char* wo = (const signed char*)d_in[13];
  const float*       so = (const float*)d_in[14];

  // output layout: out | kc | vc | new_step (flat, return order)
  float* out   = (float*)d_out;
  float* kc    = out + (size_t)B_ * T_ * D_;
  float* vc    = kc  + (size_t)B_ * M_ * HK_ * DK_;
  float* stepo = vc  + (size_t)B_ * M_ * HK_ * DK_;

  // workspace: qbuf f32 | kbuf f32 | vbuf f32 | attn bf16  (~4.2 MB)
  float*  qbuf  = (float*)d_ws;
  float*  kbuf  = qbuf + (size_t)B_ * T_ * HQ_ * DK_;
  float*  vbuf  = kbuf + (size_t)B_ * T_ * HK_ * DK_;
  __bf16* attnb = (__bf16*)(vbuf + (size_t)B_ * T_ * HK_ * DK_);

  // QKV projections (bf16 WMMA, LDS-staged dequant weights)
  gemm_w8_kernel<false><<<256, 256, 0, stream>>>(query, wq, sq, qbuf, HQ_ * DK_, D_);
  gemm_w8_kernel<false><<<64,  256, 0, stream>>>(key_,  wk, sk, kbuf, HK_ * DK_, D_);
  gemm_w8_kernel<false><<<64,  256, 0, stream>>>(value, wv, sv, vbuf, HK_ * DK_, D_);

  // RoPE
  rope_kernel<<<1024, 256, 0, stream>>>(qbuf, step, HQ_);
  rope_kernel<<<256,  256, 0, stream>>>(kbuf, step, HK_);

  // Bulk cache copy + scatter of new rows + new_step
  size_t cache_bytes = (size_t)B_ * M_ * HK_ * DK_ * sizeof(float);
  hipMemcpyAsync(kc, mem_k, cache_bytes, hipMemcpyDeviceToDevice, stream);
  hipMemcpyAsync(vc, mem_v, cache_bytes, hipMemcpyDeviceToDevice, stream);
  cache_update_kernel<<<512, 256, 0, stream>>>(kbuf, vbuf, step, kc, vc, stepo);

  // Grouped flash attention: 64 blocks x 4 waves (one q-head each)
  attn_kernel<<<64, 128, 0, stream>>>(qbuf, kc, vc, mask, step, attnb);

  // Output projection from bf16 attn
  gemm_w8_kernel<true><<<256, 256, 0, stream>>>(attnb, wo, so, out, D_, HQ_ * DK_);
}
